// JumpAttention_39599598469825
// MI455X (gfx1250) — matrix operand order repaired and checked by reading, not compiled, not run on hardware
//
#include <hip/hip_runtime.h>
#include <hip/hip_bf16.h>

typedef unsigned short u16;
typedef __attribute__((ext_vector_type(16))) __bf16 v16bf;
typedef __attribute__((ext_vector_type(8)))  float  v8f;
typedef __attribute__((__vector_size__(16))) int vi4;   // 16B transfer unit

#define B_  8
#define N_  16384
#define C_  256
#define H_  4
#define HD_ 64
#define C3_ 768

// ---------------------------------------------------------------------------
// Async global->LDS copy (CDNA5 GLOBAL_LOAD_ASYNC_TO_LDS_B128, ASYNCcnt).
// Guarded: falls back to synchronous vector copy if the builtin is absent.
// ---------------------------------------------------------------------------
#if defined(__has_builtin)
#if __has_builtin(__builtin_amdgcn_global_load_async_to_lds_b128)
#define HAVE_ASYNC_LDS 1
#endif
#endif

#ifdef HAVE_ASYNC_LDS
__device__ __forceinline__ void async_cp16(void* lds, const void* gsrc) {
  // LDS generic pointers: low 32 bits are the LDS byte offset; global generic
  // pointers are numerically identical to AS1. Cast via integers.
  __builtin_amdgcn_global_load_async_to_lds_b128(
      (__attribute__((address_space(1))) vi4*)(unsigned long long)(size_t)gsrc,
      (__attribute__((address_space(3))) vi4*)(unsigned int)(size_t)lds,
      0, 0);
}
#if __has_builtin(__builtin_amdgcn_s_wait_asynccnt)
#define WAIT_ASYNC() __builtin_amdgcn_s_wait_asynccnt(0)
#else
#define WAIT_ASYNC() asm volatile("s_wait_asynccnt 0" ::: "memory")
#endif
#else
#define WAIT_ASYNC() ((void)0)
#endif

__device__ __forceinline__ void cp16(void* lds, const void* gsrc) {
#ifdef HAVE_ASYNC_LDS
  async_cp16(lds, gsrc);
#else
  *(uint4*)lds = *(const uint4*)gsrc;
#endif
}

__device__ __forceinline__ u16 f2bf(float f) {
  unsigned u = __float_as_uint(f);
  u += 0x7FFFu + ((u >> 16) & 1u);   // round-to-nearest-even
  return (u16)(u >> 16);
}
__device__ __forceinline__ float bf2f(u16 h) {
  return __uint_as_float(((unsigned)h) << 16);
}
__device__ __forceinline__ v16bf frag_cast(uint4 lo, uint4 hi) {
  union { uint4 u[2]; v16bf v; } c;
  c.u[0] = lo; c.u[1] = hi;
  return c.v;
}
__device__ __forceinline__ v8f wmma_bf16(v16bf a, v16bf b, v8f c) {
  return __builtin_amdgcn_wmma_f32_16x16x32_bf16(false, a, false, b,
                                                 (short)0, c, false, false);
}
// A fragment (16x32 bf16, row-major tile): p = &tile[(m0+lm)*pitch + k0 + 8*lh]
__device__ __forceinline__ v16bf load_frag_a(const u16* p) {
  return frag_cast(*(const uint4*)p, *(const uint4*)(p + 16));
}
// B fragment (32x16 bf16, staged N-major): p = &tileT[(n0+lm)*pitch + k0 + 8*lh]
__device__ __forceinline__ v16bf load_frag_b(const u16* p) {
  return frag_cast(*(const uint4*)p, *(const uint4*)(p + 16));
}

// ---------------------------------------------------------------------------
// Kernel 1: qkv = x @ W  (bf16 WMMA, fp32 accum), store q/k/v bf16 channel-major
// ---------------------------------------------------------------------------
#define TN1 64
#define XP1 264   // 256 + 8 pad (u16 elems)
#define WP1 264

__global__ __launch_bounds__(256) void qkv_proj_kernel(
    const float* __restrict__ x1, const float* __restrict__ x2,
    const float* __restrict__ w,
    u16* __restrict__ qb, u16* __restrict__ kb, u16* __restrict__ vb)
{
  __shared__ __align__(16) u16 xs[TN1 * XP1];   // [token][k] bf16
  __shared__ __align__(16) u16 wt[32 * WP1];    // [n][k] bf16 (transposed W cols)

  const int tid  = threadIdx.x;
  const int lane = tid & 31;
  const int lm   = lane & 15;
  const int lh   = lane >> 4;
  const int wave = tid >> 5;

  const int n0 = blockIdx.x * TN1;
  const int b  = blockIdx.y;
  const int s  = blockIdx.z;

  const float* x = (s ? x2 : x1) + ((size_t)b * N_ + n0) * C_;

  // stage x tile -> bf16 LDS, row-major [token][k]
  for (int i = tid; i < TN1 * (C_ / 4); i += 256) {
    int r  = i >> 6;
    int c4 = (i & 63) * 4;
    float4 v = *(const float4*)(x + (size_t)r * C_ + c4);
    u16* d = &xs[r * XP1 + c4];
    d[0] = f2bf(v.x); d[1] = f2bf(v.y); d[2] = f2bf(v.z); d[3] = f2bf(v.w);
  }

  const int m0   = (wave & 3) * 16;   // token sub-tile
  const int jcol = wave >> 2;         // which of 2 staged 16-wide col tiles

  const size_t sb = (size_t)s * B_ + b;

  for (int jp = 0; jp < 24; ++jp) {
    const int jc0 = jp * 32;
    __syncthreads();
    // stage 32 W columns transposed: wt[n][k]
    {
      int nn = tid & 31;
      int kg = tid >> 5;
      for (int it = 0; it < 32; ++it) {
        int k = it * 8 + kg;
        wt[nn * WP1 + k] = f2bf(w[(size_t)k * C3_ + jc0 + nn]);
      }
    }
    __syncthreads();

    v8f acc = {};
    #pragma unroll
    for (int ks = 0; ks < 8; ++ks) {
      const int k0 = ks * 32;
      v16bf a  = load_frag_a(&xs[(m0 + lm) * XP1 + k0 + 8 * lh]);
      v16bf bm = load_frag_b(&wt[(jcol * 16 + lm) * WP1 + k0 + 8 * lh]);
      acc = wmma_bf16(a, bm, acc);
    }

    // out column = jc0 + jcol*16 + lm ; tokens = n0 + m0 + 8*lh + j
    const int jc  = jc0 + jcol * 16;
    const int typ = jc >> 8;          // 0=q,1=k,2=v
    const int ch  = jc & 255;
    const int h   = ch >> 6;
    const int d0  = ch & 63;
    u16* dst = (typ == 0) ? qb : ((typ == 1) ? kb : vb);
    const size_t row = (sb * H_ + h) * HD_ + (size_t)(d0 + lm);
    const size_t off = row * N_ + (size_t)(n0 + m0 + 8 * lh);
    uint4 pk;
    pk.x = (unsigned)f2bf(acc[0]) | ((unsigned)f2bf(acc[1]) << 16);
    pk.y = (unsigned)f2bf(acc[2]) | ((unsigned)f2bf(acc[3]) << 16);
    pk.z = (unsigned)f2bf(acc[4]) | ((unsigned)f2bf(acc[5]) << 16);
    pk.w = (unsigned)f2bf(acc[6]) | ((unsigned)f2bf(acc[7]) << 16);
    *(uint4*)(dst + off) = pk;
  }
}

// ---------------------------------------------------------------------------
// Kernel 2: per (s,b,h): S[64][64] = Q·K^T over N, plus per-channel sumsq.
// Double-buffered: async global->LDS copy of chunk i+1 overlaps WMMA on chunk i.
// ---------------------------------------------------------------------------
#define QP 72   // 64 + 8 pad

__global__ __launch_bounds__(256) void gram_kernel(
    const u16* __restrict__ qb, const u16* __restrict__ kb,
    float* __restrict__ Sb, float* __restrict__ nqb, float* __restrict__ nkb)
{
  __shared__ __align__(16) u16 qt[2][HD_ * QP];
  __shared__ __align__(16) u16 kt[2][HD_ * QP];
  __shared__ float sq[HD_], sk[HD_];

  const int tid  = threadIdx.x;
  const int lane = tid & 31;
  const int lm   = lane & 15;
  const int lh   = lane >> 4;
  const int wave = tid >> 5;

  const int h = blockIdx.x, b = blockIdx.y, s = blockIdx.z;
  const size_t g = ((size_t)s * B_ + b) * H_ + h;
  const u16* q = qb + g * HD_ * (size_t)N_;
  const u16* k = kb + g * HD_ * (size_t)N_;

  const int id0 = wave, id1 = wave + 8;           // 16 sub-tiles over 8 waves
  const int r0 = (id0 >> 2) * 16, c0 = (id0 & 3) * 16;
  const int r1 = (id1 >> 2) * 16, c1 = (id1 & 3) * 16;
  v8f acc0 = {}, acc1 = {};

  const int dch = tid & 63;
  const int seg = tid >> 6;
  float aq = 0.f, ak = 0.f;

  // stage one 64-wide K chunk (64 rows x 64 bf16) into buffer `buf`
  auto stage = [&](int buf, int ch) {
    for (int i = tid; i < HD_ * 8; i += 256) {
      int r  = i >> 3;
      int c8 = (i & 7) * 8;
      cp16(&qt[buf][r * QP + c8], q + (size_t)r * N_ + ch + c8);
      cp16(&kt[buf][r * QP + c8], k + (size_t)r * N_ + ch + c8);
    }
  };

  stage(0, 0);
  WAIT_ASYNC();
  __syncthreads();

  int p = 0;
  for (int ci = 0; ci < N_ / 64; ++ci) {
    const int ch = ci * 64;
    if (ch + 64 < N_) stage(p ^ 1, ch + 64);   // prefetch next chunk (async)

    // per-channel sum of squares on current chunk
    const u16* qrow = &qt[p][dch * QP];
    const u16* krow = &kt[p][dch * QP];
    #pragma unroll
    for (int c = seg * 16; c < seg * 16 + 16; ++c) {
      float qv = bf2f(qrow[c]); aq += qv * qv;
      float kv = bf2f(krow[c]); ak += kv * kv;
    }
    // Gram WMMAs on current chunk
    #pragma unroll
    for (int ks = 0; ks < 2; ++ks) {
      const int k0 = ks * 32;
      v16bf a0 = load_frag_a(&qt[p][(r0 + lm) * QP + k0 + 8 * lh]);
      v16bf b0 = load_frag_b(&kt[p][(c0 + lm) * QP + k0 + 8 * lh]);
      acc0 = wmma_bf16(a0, b0, acc0);
      v16bf a1 = load_frag_a(&qt[p][(r1 + lm) * QP + k0 + 8 * lh]);
      v16bf b1 = load_frag_b(&kt[p][(c1 + lm) * QP + k0 + 8 * lh]);
      acc1 = wmma_bf16(a1, b1, acc1);
    }

    WAIT_ASYNC();     // next chunk resident
    __syncthreads();  // all waves done reading current chunk
    p ^= 1;
  }

  float* S = Sb + g * (HD_ * HD_);
  #pragma unroll
  for (int j = 0; j < 8; ++j) {
    S[(r0 + j + 8 * lh) * HD_ + c0 + lm] = acc0[j];
    S[(r1 + j + 8 * lh) * HD_ + c1 + lm] = acc1[j];
  }

  if (tid < HD_) { sq[tid] = 0.f; sk[tid] = 0.f; }
  __syncthreads();
  atomicAdd(&sq[dch], aq);
  atomicAdd(&sk[dch], ak);
  __syncthreads();
  if (tid < HD_) {
    nqb[g * HD_ + tid] = sq[tid];
    nkb[g * HD_ + tid] = sk[tid];
  }
}

// ---------------------------------------------------------------------------
// Kernel 3: attn = softmax(S / (||q_c||*||k_d||) * temp) -> bf16
// ---------------------------------------------------------------------------
__global__ __launch_bounds__(64) void attn_softmax_kernel(
    const float* __restrict__ Sb, const float* __restrict__ nqb,
    const float* __restrict__ nkb, const float* __restrict__ t1,
    const float* __restrict__ t2, u16* __restrict__ ab)
{
  const int h = blockIdx.x, b = blockIdx.y, s = blockIdx.z;
  const size_t g = ((size_t)s * B_ + b) * H_ + h;
  const float* S  = Sb  + g * (HD_ * HD_);
  const float* nq = nqb + g * HD_;
  const float* nk = nkb + g * HD_;
  const float temp = (s ? t2 : t1)[h];

  __shared__ float rk[HD_];
  const int c = threadIdx.x;
  rk[c] = 1.f / fmaxf(sqrtf(nk[c]), 1e-12f);
  __syncthreads();

  const float qi = 1.f / fmaxf(sqrtf(nq[c]), 1e-12f);
  float l[HD_];
  float m = -3.402823466e38f;
  for (int d = 0; d < HD_; ++d) {
    float v = S[c * HD_ + d] * qi * rk[d] * temp;
    l[d] = v;
    m = fmaxf(m, v);
  }
  float sum = 0.f;
  for (int d = 0; d < HD_; ++d) { l[d] = __expf(l[d] - m); sum += l[d]; }
  const float inv = 1.f / sum;
  for (int d = 0; d < HD_; ++d)
    ab[g * (HD_ * HD_) + c * HD_ + d] = f2bf(l[d] * inv);
}

// ---------------------------------------------------------------------------
// Kernel 4: out[b,n,h*64+c] = (attn1 @ v2 + attn2 @ v1)  (single accumulator)
// ---------------------------------------------------------------------------
#define AP4 72
#define VP4 72

__global__ __launch_bounds__(256) void out_kernel(
    const u16* __restrict__ vb, const u16* __restrict__ ab,
    float* __restrict__ out)
{
  __shared__ __align__(16) u16 a1t[HD_ * AP4], a2t[HD_ * AP4];
  __shared__ __align__(16) u16 v1t[128 * VP4], v2t[128 * VP4];

  const int tid  = threadIdx.x;
  const int lane = tid & 31;
  const int lm   = lane & 15;
  const int lh   = lane >> 4;
  const int wave = tid >> 5;

  const int n0 = blockIdx.x * 128;
  const int b  = blockIdx.y;

  for (int h = 0; h < H_; ++h) {
    const size_t g1 = ((size_t)0 * B_ + b) * H_ + h;
    const size_t g2 = ((size_t)1 * B_ + b) * H_ + h;
    __syncthreads();
    // attn tiles: straight row-major copy -> async engine
    for (int i = tid; i < HD_ * 8; i += 256) {
      int r = i >> 3, c8 = (i & 7) * 8;
      cp16(&a1t[r * AP4 + c8], ab + g1 * (HD_*HD_) + r * HD_ + c8);
      cp16(&a2t[r * AP4 + c8], ab + g2 * (HD_*HD_) + r * HD_ + c8);
    }
    // v tiles need an in-flight transpose -> synchronous staging
    const u16* v1 = vb + g1 * HD_ * (size_t)N_;
    const u16* v2 = vb + g2 * HD_ * (size_t)N_;
    for (int i = tid; i < HD_ * 16; i += 256) {
      int d = i >> 4, c8 = (i & 15) * 8;
      uint4 p1 = *(const uint4*)(v1 + (size_t)d * N_ + n0 + c8);
      uint4 p2 = *(const uint4*)(v2 + (size_t)d * N_ + n0 + c8);
      const u16* u1 = (const u16*)&p1;
      const u16* u2 = (const u16*)&p2;
      #pragma unroll
      for (int e = 0; e < 8; ++e) {      // transpose into [token][d]
        v1t[(c8 + e) * VP4 + d] = u1[e];
        v2t[(c8 + e) * VP4 + d] = u2[e];
      }
    }
    WAIT_ASYNC();
    __syncthreads();

    #pragma unroll
    for (int sub = 0; sub < 4; ++sub) {
      const int idd = wave * 4 + sub;     // 32 sub-tiles: 4 c-tiles x 8 n-tiles
      const int ct = idd >> 3, nt = idd & 7;
      v8f acc = {};
      #pragma unroll
      for (int ks = 0; ks < 2; ++ks) {
        const int k0 = ks * 32;
        v16bf fa1 = load_frag_a(&a1t[(ct * 16 + lm) * AP4 + k0 + 8 * lh]);
        v16bf fb2 = load_frag_b(&v2t[(nt * 16 + lm) * VP4 + k0 + 8 * lh]);
        acc = wmma_bf16(fa1, fb2, acc);
        v16bf fa2 = load_frag_a(&a2t[(ct * 16 + lm) * AP4 + k0 + 8 * lh]);
        v16bf fb1 = load_frag_b(&v1t[(nt * 16 + lm) * VP4 + k0 + 8 * lh]);
        acc = wmma_bf16(fa2, fb1, acc);
      }
      const int ntok = n0 + nt * 16 + lm;
      float* o = out + ((size_t)b * N_ + ntok) * C_ + h * HD_ + ct * 16 + 8 * lh;
      float4 o0 = { acc[0], acc[1], acc[2], acc[3] };
      float4 o1 = { acc[4], acc[5], acc[6], acc[7] };
      *(float4*)o = o0;
      *(float4*)(o + 4) = o1;
    }
  }
}

// ---------------------------------------------------------------------------
extern "C" void kernel_launch(void* const* d_in, const int* in_sizes, int n_in,
                              void* d_out, int out_size, void* d_ws, size_t ws_size,
                              hipStream_t stream) {
  (void)in_sizes; (void)n_in; (void)out_size; (void)ws_size;
  const float* x1 = (const float*)d_in[0];
  const float* x2 = (const float*)d_in[1];
  const float* w  = (const float*)d_in[2];
  const float* t1 = (const float*)d_in[3];
  const float* t2 = (const float*)d_in[4];
  float* out = (float*)d_out;

  const size_t qe = (size_t)2 * B_ * C_ * N_;     // bf16 elems per q/k/v tensor
  u16* qb = (u16*)d_ws;
  u16* kb = qb + qe;
  u16* vb = kb + qe;
  float* Sb  = (float*)(vb + qe);
  float* nqb = Sb + (size_t)2 * B_ * H_ * HD_ * HD_;
  float* nkb = nqb + (size_t)2 * B_ * H_ * HD_;
  u16* ab = (u16*)(nkb + (size_t)2 * B_ * H_ * HD_);

  qkv_proj_kernel<<<dim3(N_ / TN1, B_, 2), 256, 0, stream>>>(x1, x2, w, qb, kb, vb);
  gram_kernel<<<dim3(H_, B_, 2), 256, 0, stream>>>(qb, kb, Sb, nqb, nkb);
  attn_softmax_kernel<<<dim3(H_, B_, 2), 64, 0, stream>>>(Sb, nqb, nkb, t1, t2, ab);
  out_kernel<<<dim3(N_ / 128, B_), 256, 0, stream>>>(vb, ab, out);
}
